// RNN_10539849744739
// MI455X (gfx1250) — matrix operand order repaired and checked by reading, not compiled, not run on hardware
//
#include <hip/hip_runtime.h>
#include <hip/hip_bf16.h>
#include <stdint.h>

#define HIDDEN 32

typedef __attribute__((ext_vector_type(16))) __bf16 v16bf;
typedef __attribute__((ext_vector_type(2)))  __bf16 v2bf;
typedef __attribute__((ext_vector_type(8)))  float  v8f;

union BfPack { v16bf v; uint32_t d[8]; };

// f32 pair -> packed bf16 dword.  Prefer the dedicated builtin; otherwise use
// native __bf16 fptrunc, which clang lowers to v_cvt_pk_bf16_f32 on gfx1250.
__device__ __forceinline__ uint32_t packbf(float lo, float hi) {
#if __has_builtin(__builtin_amdgcn_cvt_pk_bf16_f32)
    union { v2bf v; uint32_t u; } c;
    c.v = __builtin_amdgcn_cvt_pk_bf16_f32(lo, hi);
    return c.u;
#else
    union { v2bf v; uint32_t u; } c;
    c.v[0] = (__bf16)lo;
    c.v[1] = (__bf16)hi;
    return c.u;
#endif
}

// lane <-> lane^16 exchange, VALU-only (v_permlanex16_b32), DS fallback
__device__ __forceinline__ uint32_t xswap32(uint32_t v) {
#if __has_builtin(__builtin_amdgcn_permlanex16)
    return __builtin_amdgcn_permlanex16(v, v, 0x76543210u, 0xfedcba98u, false, false);
#else
    return (uint32_t)__builtin_amdgcn_ds_swizzle((int)v, 0x401F); // SWAPX16
#endif
}

__device__ __forceinline__ float dtanh(float x) {
#if __has_builtin(__builtin_amdgcn_tanhf)
    return __builtin_amdgcn_tanhf(x);      // v_tanh_f32 (gfx1250 TRANS op)
#else
    return tanhf(x);
#endif
}

// One wave per 16-batch tile.
//   A0/A1 = W_hh rows 0-15 / 16-31 (bf16, loop-invariant VGPRs)
//   A2    = W_out in row 0, zeros elsewhere (output projection as WMMA)
//   B     = h_t^T (bf16), rebuilt each step with packed permlanex16 swaps
//   C     = x_t*W_ih + b_ih + b_hh (f32, v_pk_fma)
__global__ __launch_bounds__(32)
void rnn_scan_wmma(const float* __restrict__ x,
                   const float* __restrict__ h0p,
                   const float* __restrict__ wih,
                   const float* __restrict__ whh,
                   const float* __restrict__ bih,
                   const float* __restrict__ bhh,
                   const float* __restrict__ wo,
                   const float* __restrict__ bo,
                   float* __restrict__ out,
                   float* __restrict__ hlast,
                   int B, int T)
{
    const int  lane = threadIdx.x;     // 0..31 (wave32)
    const int  col  = lane & 15;       // batch column inside tile
    const bool low  = lane < 16;
    const int  b    = blockIdx.x * 16 + col;
    const int  h0   = low ? 0 : 8;     // C/D row base for this lane-half

    // ---- A tiles (contiguous K layout: lanes 0-15 V0..7=K0..15;
    //                                    lanes 16-31 V0..7=K16..31)
    BfPack a0, a1, a2;
    {
        const int r  = col;
        const int kb = low ? 0 : 16;
#pragma unroll
        for (int j = 0; j < 4; ++j) {
            const int k = kb + 2 * j;
            a0.d[j]     = packbf(whh[r * HIDDEN + k],             whh[r * HIDDEN + k + 1]);
            a0.d[4 + j] = packbf(whh[r * HIDDEN + 8 + k],         whh[r * HIDDEN + 8 + k + 1]);
            a1.d[j]     = packbf(whh[(r + 16) * HIDDEN + k],      whh[(r + 16) * HIDDEN + k + 1]);
            a1.d[4 + j] = packbf(whh[(r + 16) * HIDDEN + 8 + k],  whh[(r + 16) * HIDDEN + 8 + k + 1]);
            a2.d[j]     = (r == 0) ? packbf(wo[k],      wo[k + 1])      : 0u;
            a2.d[4 + j] = (r == 0) ? packbf(wo[8 + k],  wo[8 + k + 1])  : 0u;
        }
    }

    // ---- per-lane constants
    float wih0[8], wih1[8], bb0[8], bb1[8];
#pragma unroll
    for (int m = 0; m < 8; ++m) {
        wih0[m] = wih[h0 + m];
        wih1[m] = wih[16 + h0 + m];
        bb0[m]  = bih[h0 + m]      + bhh[h0 + m];
        bb1[m]  = bih[16 + h0 + m] + bhh[16 + h0 + m];
    }
    // C operand for the output-projection WMMA: b_out pre-folded into row 0
    v8f cout_init = {};
    cout_init[0] = bo[0];

    // ---- initial B operand (h0^T): lanes 0-15 K=0..15, lanes 16-31 K=16..31
    BfPack bm;
    {
        const int kb = low ? 0 : 16;
#pragma unroll
        for (int v = 0; v < 8; ++v) {
            const int k = kb + 2 * v;
            bm.d[v] = packbf(h0p[b * HIDDEN + k], h0p[b * HIDDEN + k + 1]);
        }
    }

    float t0f[8], t1f[8];
#pragma unroll
    for (int m = 0; m < 8; ++m) { t0f[m] = 0.f; t1f[m] = 0.f; }

    for (int tt = 0; tt < T; tt += 4) {
        const float4 xq = *(const float4*)(x + (size_t)b * T + tt);
        float4 ob;
#pragma unroll
        for (int s = 0; s < 4; ++s) {
            const float xv = (s == 0) ? xq.x : (s == 1) ? xq.y : (s == 2) ? xq.z : xq.w;

            // C = xp_t tile (input projection + biases), f32, D-layout
            v8f c0, c1;
#pragma unroll
            for (int m = 0; m < 8; ++m) {
                c0[m] = fmaf(xv, wih0[m], bb0[m]);
                c1[m] = fmaf(xv, wih1[m], bb1[m]);
            }

            // D = W_hh * h^T + xp   (two independent 16x16x32 bf16 WMMAs)
            c0 = __builtin_amdgcn_wmma_f32_16x16x32_bf16(false, a0.v, false, bm.v,
                                                         (short)0, c0, false, false);
            c1 = __builtin_amdgcn_wmma_f32_16x16x32_bf16(false, a1.v, false, bm.v,
                                                         (short)0, c1, false, false);

            // h_{t+1} = tanh(D)  (v_tanh_f32, TRANS co-execution)
#pragma unroll
            for (int m = 0; m < 8; ++m) {
                t0f[m] = dtanh(c0[m]);
                t1f[m] = dtanh(c1[m]);
            }

            // Rebuild B operand: pack to bf16 pairs (v_cvt_pk_bf16_f32), then
            // swap packed dwords across lane halves (permlanex16, no LDS).
#pragma unroll
            for (int v = 0; v < 4; ++v) {
                const uint32_t pt0 = packbf(t0f[2 * v], t0f[2 * v + 1]); // rows 2v,2v+1   (+h0)
                const uint32_t pt1 = packbf(t1f[2 * v], t1f[2 * v + 1]); // rows 16+2v,... (+h0)
                const uint32_t ps  = low ? pt1 : pt0;    // what the *other* half needs
                const uint32_t pw  = xswap32(ps);
                bm.d[v]     = low ? pt0 : pw;            // K 0..7   | K 16..23
                bm.d[4 + v] = low ? pw  : pt1;           // K 8..15  | K 24..31
            }

            // out_t = W_out * h_t + b_out as a 3rd WMMA (row 0 of a2),
            // off the recurrence critical path.
            v8f co = __builtin_amdgcn_wmma_f32_16x16x32_bf16(false, a2.v, false, bm.v,
                                                             (short)0, cout_init,
                                                             false, false);
            ((float*)&ob)[s] = co[0];   // valid on lanes 0-15 (row 0)
        }
        if (low) *(float4*)(out + (size_t)b * T + tt) = ob;
    }

    // final hidden state, [B, H] f32
#pragma unroll
    for (int m = 0; m < 8; ++m) {
        hlast[b * HIDDEN + h0 + m]      = t0f[m];
        hlast[b * HIDDEN + 16 + h0 + m] = t1f[m];
    }
}

extern "C" void kernel_launch(void* const* d_in, const int* in_sizes, int n_in,
                              void* d_out, int out_size, void* d_ws, size_t ws_size,
                              hipStream_t stream) {
    const float* x   = (const float*)d_in[0];
    const float* h0p = (const float*)d_in[1];
    const float* wih = (const float*)d_in[2];
    const float* whh = (const float*)d_in[3];
    const float* bih = (const float*)d_in[4];
    const float* bhh = (const float*)d_in[5];
    const float* wo  = (const float*)d_in[6];
    const float* bo  = (const float*)d_in[7];

    const int B = in_sizes[1] / HIDDEN;   // h_state: 1*B*H
    const int T = in_sizes[0] / B;        // x: B*T*1

    float* out   = (float*)d_out;
    float* hlast = out + (size_t)B * T;

    rnn_scan_wmma<<<dim3(B / 16), dim3(32), 0, stream>>>(
        x, h0p, wih, whh, bih, bhh, wo, bo, out, hlast, B, T);
}
